// MyGraphTransformer_20005957664852
// MI455X (gfx1250) — matrix operand order repaired and checked by reading, not compiled, not run on hardware
//
#include <hip/hip_runtime.h>
#include <hip/hip_bf16.h>
#include <math.h>

typedef _Float16 f16;
typedef __attribute__((ext_vector_type(16))) _Float16 v16h;
typedef __attribute__((ext_vector_type(8)))  float    v8f;

#define NN 50000
#define EE 800000
#define DD 128
#define HH 4
#define CC 32
#define LL 3

__device__ __forceinline__ void atomicMaxFloat(float* addr, float val) {
    unsigned int bits = __float_as_uint(val);
    if (bits & 0x80000000u) {
        atomicMin((unsigned int*)addr, bits);       // negative: unsigned min == float max
    } else {
        atomicMax((int*)addr, (int)bits);           // non-negative: signed max == float max
    }
}

// ---------------------------------------------------------------------------
// Pack Wq|Wk|Wv|Ws (each [128 x 128], row-major in-major=K) into f16,
// transposed to [outcol][k] so B fragments are contiguous per column.
// ---------------------------------------------------------------------------
__global__ __launch_bounds__(256) void convert_w_kernel(
    const float* __restrict__ Wq, const float* __restrict__ Wk,
    const float* __restrict__ Wv, const float* __restrict__ Ws,
    f16* __restrict__ whT)
{
    int t = blockIdx.x * 256 + threadIdx.x;      // 0 .. 512*128-1
    int nc = t >> 7;                             // concatenated out column 0..511
    int kk = t & 127;                            // K index
    const float* W = (nc < 128) ? Wq : (nc < 256) ? Wk : (nc < 384) ? Wv : Ws;
    int j = nc & 127;
    whT[(size_t)nc * DD + kk] = (f16)W[(size_t)kk * DD + j];
}

// ---------------------------------------------------------------------------
// Fused Q/K/V/S GEMM: x[50000x128] @ [Wq|Wk|Wv|Ws] via v_wmma_f32_16x16x32_f16.
// Block = 256 threads (8 waves) handles 16 rows x 512 cols; each wave owns
// 4 column tiles of 16. S epilogue folds in bias + residual x.
// ---------------------------------------------------------------------------
__global__ __launch_bounds__(256) void fused_qkvs_wmma_kernel(
    const float* __restrict__ x, const f16* __restrict__ whT,
    const float* __restrict__ bq, const float* __restrict__ bk,
    const float* __restrict__ bv, const float* __restrict__ bs,
    float* __restrict__ q, float* __restrict__ k,
    float* __restrict__ v, float* __restrict__ s)
{
    const int lane = threadIdx.x & 31;
    const int wave = threadIdx.x >> 5;
    const int m0   = blockIdx.x << 4;     // N = 50000 = 3125 * 16, exact
    const int lrow = lane & 15;
    const int hi   = lane >> 4;           // 0 or 1: K-half selector

    v8f acc[4] = {};

    const float* xrow = x + (size_t)(m0 + lrow) * DD;

    #pragma unroll
    for (int ks = 0; ks < 4; ++ks) {
        // A fragment (16x32 f16): lane holds row lrow; K = hi*16 + j, contiguous
        // (matches B fragment k order)
        const float* ap = xrow + ks * 32 + hi * 16;
        float4 a0 = *(const float4*)(ap);
        float4 a1 = *(const float4*)(ap + 4);
        float4 a2 = *(const float4*)(ap + 8);
        float4 a3 = *(const float4*)(ap + 12);
        v16h A;
        A[0]=(f16)a0.x;  A[1]=(f16)a0.y;  A[2]=(f16)a0.z;  A[3]=(f16)a0.w;
        A[4]=(f16)a1.x;  A[5]=(f16)a1.y;  A[6]=(f16)a1.z;  A[7]=(f16)a1.w;
        A[8]=(f16)a2.x;  A[9]=(f16)a2.y;  A[10]=(f16)a2.z; A[11]=(f16)a2.w;
        A[12]=(f16)a3.x; A[13]=(f16)a3.y; A[14]=(f16)a3.z; A[15]=(f16)a3.w;

        #pragma unroll
        for (int t = 0; t < 4; ++t) {
            const int n0 = ((wave << 2) + t) << 4;
            // B fragment (32x16 f16): lane holds col lrow; K = hi*16 + j, contiguous
            const f16* bp = whT + (size_t)(n0 + lrow) * DD + ks * 32 + hi * 16;
            v16h B = *(const v16h*)bp;
            acc[t] = __builtin_amdgcn_wmma_f32_16x16x32_f16(
                false, A, false, B, (short)0, acc[t], false, false);
        }
    }

    // Epilogue: C/D layout -> VGPR r holds row m0 + r + 8*hi, col = n0 + lrow
    #pragma unroll
    for (int t = 0; t < 4; ++t) {
        const int n0   = ((wave << 2) + t) << 4;
        const int m    = n0 >> 7;                 // 0=Q 1=K 2=V 3=S
        const int jcol = (n0 & 127) + lrow;
        const float* bias = (m == 0) ? bq : (m == 1) ? bk : (m == 2) ? bv : bs;
        float*       outb = (m == 0) ? q  : (m == 1) ? k  : (m == 2) ? v  : s;
        const float bval = bias[jcol];
        #pragma unroll
        for (int r = 0; r < 8; ++r) {
            const int row = m0 + r + (hi << 3);
            float val = acc[t][r] + bval;
            if (m == 3) val += x[(size_t)row * DD + jcol];   // +x residual into S
            outb[(size_t)row * DD + jcol] = val;
        }
    }
}

// ---------------------------------------------------------------------------
// Init per-(node,head) softmax stats.
// ---------------------------------------------------------------------------
__global__ __launch_bounds__(256) void init_stats_kernel(float* __restrict__ amax,
                                                         float* __restrict__ denom)
{
    int t = blockIdx.x * 256 + threadIdx.x;
    if (t < NN * HH) {
        amax[t]  = -INFINITY;
        denom[t] = 0.0f;
    }
}

// ---------------------------------------------------------------------------
// Pass 1: per-(edge,head) attention logit + segment max over dst.
// ---------------------------------------------------------------------------
__global__ __launch_bounds__(256) void edge_logits_kernel(
    const int* __restrict__ srcp, const int* __restrict__ dstp,
    const float* __restrict__ q, const float* __restrict__ k,
    float* __restrict__ ex, float* __restrict__ amax)
{
    int t = blockIdx.x * 256 + threadIdx.x;     // E*H = 3.2M, exact grid
    int e = t >> 2, h = t & 3;
    int si = srcp[e], di = dstp[e];
    const float4* qp = (const float4*)(q + (size_t)di * DD + h * CC);
    const float4* kp = (const float4*)(k + (size_t)si * DD + h * CC);
    float dot = 0.0f;
    #pragma unroll
    for (int i = 0; i < 8; ++i) {
        float4 a = qp[i], b = kp[i];
        dot += a.x * b.x + a.y * b.y + a.z * b.z + a.w * b.w;
    }
    float alpha = dot * 0.17677669529663687f;   // 1/sqrt(32)
    ex[t] = alpha;
    atomicMaxFloat(&amax[di * HH + h], alpha);
}

// ---------------------------------------------------------------------------
// Pass 2: exp(logit - max) + segment sum over dst.
// ---------------------------------------------------------------------------
__global__ __launch_bounds__(256) void edge_exp_kernel(
    const int* __restrict__ dstp, float* __restrict__ ex,
    const float* __restrict__ amax, float* __restrict__ denom)
{
    int t = blockIdx.x * 256 + threadIdx.x;
    int e = t >> 2, h = t & 3;
    int di = dstp[e];
    float w = expf(ex[t] - amax[di * HH + h]);
    ex[t] = w;
    atomicAdd(&denom[di * HH + h], w);
}

// ---------------------------------------------------------------------------
// Pass 3: weighted scatter of v[src] into s[dst] (s already holds x@Ws+bs+x).
// ---------------------------------------------------------------------------
__global__ __launch_bounds__(256) void edge_aggregate_kernel(
    const int* __restrict__ srcp, const int* __restrict__ dstp,
    const float* __restrict__ v, const float* __restrict__ ex,
    const float* __restrict__ denom, float* __restrict__ s)
{
    int t = blockIdx.x * 256 + threadIdx.x;
    int e = t >> 2, h = t & 3;
    int si = srcp[e], di = dstp[e];
    float w = ex[t] / (denom[di * HH + h] + 1e-16f);
    const float4* vp = (const float4*)(v + (size_t)si * DD + h * CC);
    float* op = s + (size_t)di * DD + h * CC;
    #pragma unroll
    for (int i = 0; i < 8; ++i) {
        float4 a = vp[i];
        atomicAdd(op + 4 * i + 0, a.x * w);
        atomicAdd(op + 4 * i + 1, a.y * w);
        atomicAdd(op + 4 * i + 2, a.z * w);
        atomicAdd(op + 4 * i + 3, a.w * w);
    }
}

// ---------------------------------------------------------------------------
// LayerNorm (+exact GELU except last layer). One wave32 per row, 4 cols/lane.
// ---------------------------------------------------------------------------
__global__ __launch_bounds__(256) void layernorm_kernel(
    const float* __restrict__ s, const float* __restrict__ g,
    const float* __restrict__ b, float* __restrict__ out, int apply_gelu)
{
    const int lane = threadIdx.x & 31;
    const int wave = threadIdx.x >> 5;
    const int row  = (blockIdx.x << 3) + wave;   // 50000 = 6250 * 8, exact

    float4 val = *(const float4*)(s + (size_t)row * DD + lane * 4);
    float sum = val.x + val.y + val.z + val.w;
    #pragma unroll
    for (int off = 16; off; off >>= 1) sum += __shfl_xor(sum, off, 32);
    float mu = sum * (1.0f / 128.0f);

    float4 d;
    d.x = val.x - mu; d.y = val.y - mu; d.z = val.z - mu; d.w = val.w - mu;
    float vs = d.x * d.x + d.y * d.y + d.z * d.z + d.w * d.w;
    #pragma unroll
    for (int off = 16; off; off >>= 1) vs += __shfl_xor(vs, off, 32);
    float inv = rsqrtf(vs * (1.0f / 128.0f) + 1e-5f);

    float4 gg = *(const float4*)(g + lane * 4);
    float4 bb = *(const float4*)(b + lane * 4);
    float4 y;
    y.x = d.x * inv * gg.x + bb.x;
    y.y = d.y * inv * gg.y + bb.y;
    y.z = d.z * inv * gg.z + bb.z;
    y.w = d.w * inv * gg.w + bb.w;
    if (apply_gelu) {
        const float r2 = 0.70710678118654752f;
        y.x = 0.5f * y.x * (1.0f + erff(y.x * r2));
        y.y = 0.5f * y.y * (1.0f + erff(y.y * r2));
        y.z = 0.5f * y.z * (1.0f + erff(y.z * r2));
        y.w = 0.5f * y.w * (1.0f + erff(y.w * r2));
    }
    *(float4*)(out + (size_t)row * DD + lane * 4) = y;
}

// ---------------------------------------------------------------------------
extern "C" void kernel_launch(void* const* d_in, const int* in_sizes, int n_in,
                              void* d_out, int out_size, void* d_ws, size_t ws_size,
                              hipStream_t stream)
{
    (void)in_sizes; (void)n_in; (void)out_size; (void)ws_size;

    const float* x0  = (const float*)d_in[0];
    const int*   ei  = (const int*)d_in[1];
    const int*   srcp = ei;
    const int*   dstp = ei + EE;

    const size_t ND = (size_t)NN * DD;   // 6,400,000
    const size_t EH = (size_t)EE * HH;   // 3,200,000
    const size_t NH = (size_t)NN * HH;   //   200,000

    float* q     = (float*)d_ws;
    float* k     = q + ND;
    float* v     = k + ND;
    float* s     = v + ND;
    float* ex    = s + ND;
    float* amax  = ex + EH;
    float* denom = amax + NH;
    float* xbuf  = denom + NH;
    f16*   whT   = (f16*)(xbuf + ND);    // 512*128 f16; 32B-aligned offset

    const int gW    = (512 * 128) / 256;   // 256
    const int gGemm = NN / 16;             // 3125
    const int gInit = (int)((NH + 255) / 256);
    const int gEdge = (int)(EH / 256);     // 12500, exact
    const int gLN   = NN / 8;              // 6250, exact

    for (int l = 0; l < LL; ++l) {
        const int base = 2 + 10 * l;
        const float* Wq  = (const float*)d_in[base + 0];
        const float* bq  = (const float*)d_in[base + 1];
        const float* Wk  = (const float*)d_in[base + 2];
        const float* bk  = (const float*)d_in[base + 3];
        const float* Wv  = (const float*)d_in[base + 4];
        const float* bv  = (const float*)d_in[base + 5];
        const float* Ws  = (const float*)d_in[base + 6];
        const float* bs  = (const float*)d_in[base + 7];
        const float* lng = (const float*)d_in[base + 8];
        const float* lnb = (const float*)d_in[base + 9];

        const float* xin  = (l == 0) ? x0 : xbuf;
        float*       xout = (l == LL - 1) ? (float*)d_out : xbuf;

        convert_w_kernel<<<gW, 256, 0, stream>>>(Wq, Wk, Wv, Ws, whT);
        fused_qkvs_wmma_kernel<<<gGemm, 256, 0, stream>>>(
            xin, whT, bq, bk, bv, bs, q, k, v, s);
        init_stats_kernel<<<gInit, 256, 0, stream>>>(amax, denom);
        edge_logits_kernel<<<gEdge, 256, 0, stream>>>(srcp, dstp, q, k, ex, amax);
        edge_exp_kernel<<<gEdge, 256, 0, stream>>>(dstp, ex, amax, denom);
        edge_aggregate_kernel<<<gEdge, 256, 0, stream>>>(srcp, dstp, v, ex, denom, s);
        layernorm_kernel<<<gLN, 256, 0, stream>>>(s, lng, lnb, xout, l < LL - 1 ? 1 : 0);
    }
}
